// Linear_TemporalAttention_11897059410212
// MI455X (gfx1250) — compile-verified, matrix operand order checked by reading, not run on hardware
//
#include <hip/hip_runtime.h>
#include <hip/hip_bf16.h>

// Linear temporal attention, MI455X (gfx1250, wave32, WMMA).
// qkv GEMM (bf16 wmma) -> per-pixel 32x32x32 attention (bf16 wmma)
//   -> out GEMM + fused LayerNorm. Fragments loaded as 2x ds_load_b128.

typedef __bf16 bf16_t;
typedef bf16_t v16bf __attribute__((ext_vector_type(16)));
typedef float  v8f   __attribute__((ext_vector_type(8)));
typedef unsigned int uint2v __attribute__((ext_vector_type(2)));
typedef unsigned int uint4v __attribute__((ext_vector_type(4)));
typedef int v2i __attribute__((ext_vector_type(2)));

union Frag { v16bf v; uint4v q[2]; unsigned short u[16]; };

#if __has_builtin(__builtin_amdgcn_global_store_async_from_lds_b64)
#define USE_ASYNC_STORE 1
typedef __attribute__((address_space(1))) v2i glb_v2i;
typedef __attribute__((address_space(3))) v2i lds_v2i;
#endif

// ---- f32 <-> bf16 conversion --------------------------------------------
__device__ __forceinline__ unsigned short f2b_sw(float f) {
    union { float f; unsigned u; } x; x.f = f;
    unsigned u = x.u;
    u += 0x7FFFu + ((u >> 16) & 1u);   // round-to-nearest-even
    return (unsigned short)(u >> 16);
}
#if __has_builtin(__builtin_amdgcn_cvt_pk_bf16_f32)
typedef bf16_t v2bf __attribute__((ext_vector_type(2)));
__device__ __forceinline__ unsigned pack2(float a, float b) {
    union { v2bf v; unsigned u; } x;
    x.v = __builtin_amdgcn_cvt_pk_bf16_f32(a, b);
    return x.u;
}
#else
__device__ __forceinline__ unsigned pack2(float a, float b) {
    return (unsigned)f2b_sw(a) | ((unsigned)f2b_sw(b) << 16);
}
#endif
__device__ __forceinline__ unsigned short f2b(float f) {
    return (unsigned short)(pack2(f, 0.0f) & 0xFFFFu);
}
__device__ __forceinline__ float b2f(unsigned short h) {
    union { unsigned u; float f; } x; x.u = ((unsigned)h) << 16;
    return x.f;
}
__device__ __forceinline__ uint2v pack4(float a, float b, float c, float d) {
    uint2v p;
    p.x = pack2(a, b);
    p.y = pack2(c, d);
    return p;
}
// CDNA5 16-bit operand: lane group kg=(lane>>4)*8; element i covers
// K = kg+i (i<8) and K = 16+kg+(i-8) (i>=8)  ->  two contiguous 16B runs.
__device__ __forceinline__ void load_frag(Frag& f, const unsigned short* row, int lane) {
    int kg = (lane >> 4) << 3;
    f.q[0] = *(const uint4v*)(row + kg);
    f.q[1] = *(const uint4v*)(row + 16 + kg);
}

// -------------------------------------------------------------------------
// Kernel 1: qkv[bt, 0..767, n] = w_qkv(768x128) @ x[bt](128x1024), bf16 out.
// Grid: (ntile=8, mtile=6, bt=128), 256 threads.
// -------------------------------------------------------------------------
__global__ __launch_bounds__(256) void qkv_gemm(
    const float* __restrict__ x, const float* __restrict__ w_qkv,
    unsigned short* __restrict__ qkv) {
    __shared__ __align__(16) unsigned short sA[128][40];    // [m][k]
    __shared__ __align__(16) unsigned short sBt[128][40];   // [n][k] (transposed)

    const int tid = threadIdx.x;
    const int bt = blockIdx.z, m0 = blockIdx.y * 128, n0 = blockIdx.x * 128;
    const int wave = tid >> 5, lane = tid & 31;
    const int wm = wave >> 1, wn = wave & 1;     // 4x2 wave grid, 32x64 per wave
    const float oscale = (m0 >= 512) ? (1.0f / 1024.0f) : 1.0f;   // v / N

    const v8f z = {};
    v8f acc[2][4];
    for (int tm = 0; tm < 2; ++tm)
        for (int tn = 0; tn < 4; ++tn) acc[tm][tn] = z;

    for (int kk = 0; kk < 128; kk += 32) {
        // A: w_qkv 128x32, float4 loads -> packed b64 LDS stores
        for (int j = tid; j < 1024; j += 256) {
            int m = j >> 3, k4 = (j & 7) << 2;
            float4 f = *(const float4*)(w_qkv + (m0 + m) * 128 + kk + k4);
            *(uint2v*)&sA[m][k4] = pack4(f.x, f.y, f.z, f.w);
        }
        // B: x 32x128 (coalesced float4 on n) -> transposed scatter
        for (int j = tid; j < 1024; j += 256) {
            int k = j >> 5, n4 = (j & 31) << 2;
            const float* src = x + (size_t)(bt * 128 + kk + k) * 1024 + n0 + n4;
            if (kk + 32 < 128) __builtin_prefetch(src + 32 * 1024, 0, 1);
            float4 f = *(const float4*)src;
            sBt[n4 + 0][k] = f2b(f.x);
            sBt[n4 + 1][k] = f2b(f.y);
            sBt[n4 + 2][k] = f2b(f.z);
            sBt[n4 + 3][k] = f2b(f.w);
        }
        __syncthreads();

        Frag a[2], b[4];
        for (int tm = 0; tm < 2; ++tm)
            load_frag(a[tm], &sA[wm * 32 + tm * 16 + (lane & 15)][0], lane);
        for (int tn = 0; tn < 4; ++tn)
            load_frag(b[tn], &sBt[wn * 64 + tn * 16 + (lane & 15)][0], lane);
        for (int tm = 0; tm < 2; ++tm)
            for (int tn = 0; tn < 4; ++tn)
                acc[tm][tn] = __builtin_amdgcn_wmma_f32_16x16x32_bf16(
                    false, a[tm].v, false, b[tn].v, (short)0, acc[tm][tn], false, false);
        __syncthreads();
    }

    for (int tm = 0; tm < 2; ++tm)
        for (int tn = 0; tn < 4; ++tn)
            for (int v = 0; v < 8; ++v) {
                int row = m0 + wm * 32 + tm * 16 + v + ((lane >> 4) << 3);
                int col = n0 + wn * 64 + tn * 16 + (lane & 15);
                qkv[(size_t)(bt * 768 + row) * 1024 + col] = f2b(acc[tm][tn][v] * oscale);
            }
}

// -------------------------------------------------------------------------
// Kernel 2: per (b,h,n): softmax(K over t), softmax(Q over d)*scale,
//   context = Ksm @ V^T, out = context^T @ Qsm. One wave per n, 4 n / block.
// Grid: (nchunk=256, bh=32), 128 threads.
// -------------------------------------------------------------------------
__global__ __launch_bounds__(128) void attn_core(
    const unsigned short* __restrict__ qkv, unsigned short* __restrict__ att) {
    __shared__ __align__(16) unsigned short sQt[4][32][40];  // [n][t][d] (transposed)
    __shared__ __align__(16) unsigned short sK[4][32][40];   // [n][d][t]
    __shared__ __align__(16) unsigned short sV[4][32][40];   // [n][e][t]
    __shared__ __align__(16) unsigned short sCt[4][32][40];  // context^T [n][e][d]
    __shared__ __align__(16) unsigned short sOut[32][32][4]; // [e][t][nl]

    const int tid = threadIdx.x;
    const int bh = blockIdx.y, b = bh >> 3, h = bh & 7;
    const int n0 = blockIdx.x * 4;

    // load q,k,v tiles: 8-byte global loads over 4 consecutive n
    for (int j = tid; j < 3 * 32 * 32; j += 128) {
        int d = j & 31, t = (j >> 5) & 31, which = j >> 10;
        uint2v vv = *(const uint2v*)(
            qkv + (size_t)((b * 32 + t) * 768 + which * 256 + h * 32 + d) * 1024 + n0);
        union { uint2v v; unsigned short s[4]; } u; u.v = vv;
        for (int nl = 0; nl < 4; ++nl) {
            if (which == 0)      sQt[nl][t][d] = u.s[nl];
            else if (which == 1) sK[nl][d][t]  = u.s[nl];
            else                 sV[nl][d][t]  = u.s[nl];
        }
    }
    __syncthreads();

    const int wave = tid >> 5, lane = tid & 31;
    const int nl = wave;   // one n per wave

    // K softmax over t (per row d): lane = t, cross-lane reductions.
    for (int d = 0; d < 32; ++d) {
        float kv = b2f(sK[nl][d][lane]);
        float mx = kv;
        for (int off = 16; off; off >>= 1) mx = fmaxf(mx, __shfl_xor(mx, off, 32));
        float e = __expf(kv - mx);
        float s = e;
        for (int off = 16; off; off >>= 1) s += __shfl_xor(s, off, 32);
        sK[nl][d][lane] = f2b(e / s);
    }
    // Q softmax over d (per column t): lane = t owns row sQt[t][*]; * dim_head^-0.5
    {
        float mx = -3.4e38f;
        for (int d = 0; d < 32; ++d) mx = fmaxf(mx, b2f(sQt[nl][lane][d]));
        float s = 0.f;
        for (int d = 0; d < 32; ++d) s += __expf(b2f(sQt[nl][lane][d]) - mx);
        float inv = 0.17677669529663687f / s;      // (1/sqrt(32)) / s
        for (int d = 0; d < 32; ++d)
            sQt[nl][lane][d] = f2b(__expf(b2f(sQt[nl][lane][d]) - mx) * inv);
    }
    __syncthreads();

    const v8f z = {};
    // context[d,e] = sum_t Ksm[d,t] * V[e,t]
    Frag a[2], bb[2];
    for (int tm = 0; tm < 2; ++tm) load_frag(a[tm],  &sK[nl][tm * 16 + (lane & 15)][0], lane);
    for (int tn = 0; tn < 2; ++tn) load_frag(bb[tn], &sV[nl][tn * 16 + (lane & 15)][0], lane);
    v8f ctx[2][2];
    for (int tm = 0; tm < 2; ++tm)
        for (int tn = 0; tn < 2; ++tn)
            ctx[tm][tn] = __builtin_amdgcn_wmma_f32_16x16x32_bf16(
                false, a[tm].v, false, bb[tn].v, (short)0, z, false, false);
    // write context transposed [e][d] so the next A operand is K-contiguous
    for (int tm = 0; tm < 2; ++tm)
        for (int tn = 0; tn < 2; ++tn)
            for (int v = 0; v < 8; ++v)
                sCt[nl][tn * 16 + (lane & 15)][tm * 16 + v + ((lane >> 4) << 3)] =
                    f2b(ctx[tm][tn][v]);
    __syncthreads();

    // out[e,t] = sum_d context[d,e] * Qsm[d,t]
    Frag a2[2], b2[2];
    for (int tm = 0; tm < 2; ++tm) load_frag(a2[tm], &sCt[nl][tm * 16 + (lane & 15)][0], lane);
    for (int tn = 0; tn < 2; ++tn) load_frag(b2[tn], &sQt[nl][tn * 16 + (lane & 15)][0], lane);
    v8f o[2][2];
    for (int tm = 0; tm < 2; ++tm)
        for (int tn = 0; tn < 2; ++tn)
            o[tm][tn] = __builtin_amdgcn_wmma_f32_16x16x32_bf16(
                false, a2[tm].v, false, b2[tn].v, (short)0, z, false, false);

#ifdef USE_ASYNC_STORE
    // stage [e][t][nl] (nl contiguous -> 8B runs), then async LDS->global DMA
    for (int tm = 0; tm < 2; ++tm)
        for (int tn = 0; tn < 2; ++tn)
            for (int v = 0; v < 8; ++v)
                sOut[tm * 16 + v + ((lane >> 4) << 3)][tn * 16 + (lane & 15)][nl] =
                    f2b(o[tm][tn][v]);
    __syncthreads();
    for (int j = tid; j < 1024; j += 128) {
        int e = j >> 5, t = j & 31;
        size_t gidx = (size_t)((b * 32 + t) * 256 + h * 32 + e) * 1024 + n0;
        __builtin_amdgcn_global_store_async_from_lds_b64(
            (glb_v2i*)(att + gidx), (lds_v2i*)&sOut[e][t][0], 0, 0);
    }
    // ASYNCcnt drained by S_ENDPGM implicit wait-idle.
#else
    (void)sOut;
    for (int tm = 0; tm < 2; ++tm)
        for (int tn = 0; tn < 2; ++tn)
            for (int v = 0; v < 8; ++v) {
                int e = tm * 16 + v + ((lane >> 4) << 3);
                int t = tn * 16 + (lane & 15);
                att[(size_t)((b * 32 + t) * 256 + h * 32 + e) * 1024 + n0 + nl] =
                    f2b(o[tm][tn][v]);
            }
#endif
}

// -------------------------------------------------------------------------
// Kernel 3: y = LN_C( w_out(128x256) @ att[bt](256x1024) + b_out ) * g
// Grid: (ntile=8, bt=128), 256 threads.
// -------------------------------------------------------------------------
__global__ __launch_bounds__(256) void out_gemm_ln(
    const unsigned short* __restrict__ att, const float* __restrict__ w_out,
    const float* __restrict__ b_out, const float* __restrict__ g,
    float* __restrict__ y) {
    __shared__ __align__(16) unsigned short sA[128][40];    // [m][k]
    __shared__ __align__(16) unsigned short sBt[128][40];   // [n][k]
    __shared__ float pSum[4][128], pSq[4][128];
    __shared__ float sMean[128], sRstd[128];

    const int tid = threadIdx.x;
    const int bt = blockIdx.y, n0 = blockIdx.x * 128;
    const int wave = tid >> 5, lane = tid & 31;
    const int wm = wave >> 1, wn = wave & 1;

    const v8f z = {};
    v8f acc[2][4];
    for (int tm = 0; tm < 2; ++tm)
        for (int tn = 0; tn < 4; ++tn) acc[tm][tn] = z;

    for (int kk = 0; kk < 256; kk += 32) {
        for (int j = tid; j < 1024; j += 256) {
            int m = j >> 3, k4 = (j & 7) << 2;
            float4 f = *(const float4*)(w_out + m * 256 + kk + k4);
            *(uint2v*)&sA[m][k4] = pack4(f.x, f.y, f.z, f.w);
        }
        for (int j = tid; j < 1024; j += 256) {
            int k = j >> 5, n4 = (j & 31) << 2;
            uint2v vv = *(const uint2v*)(att + (size_t)(bt * 256 + kk + k) * 1024 + n0 + n4);
            union { uint2v v; unsigned short s[4]; } u; u.v = vv;
            sBt[n4 + 0][k] = u.s[0];
            sBt[n4 + 1][k] = u.s[1];
            sBt[n4 + 2][k] = u.s[2];
            sBt[n4 + 3][k] = u.s[3];
        }
        __syncthreads();

        Frag a[2], b[4];
        for (int tm = 0; tm < 2; ++tm)
            load_frag(a[tm], &sA[wm * 32 + tm * 16 + (lane & 15)][0], lane);
        for (int tn = 0; tn < 4; ++tn)
            load_frag(b[tn], &sBt[wn * 64 + tn * 16 + (lane & 15)][0], lane);
        for (int tm = 0; tm < 2; ++tm)
            for (int tn = 0; tn < 4; ++tn)
                acc[tm][tn] = __builtin_amdgcn_wmma_f32_16x16x32_bf16(
                    false, a[tm].v, false, b[tn].v, (short)0, acc[tm][tn], false, false);
        __syncthreads();
    }

    // bias + per-column partial sums (LayerNorm over the 128 rows)
    float cs[4] = {0, 0, 0, 0}, cq[4] = {0, 0, 0, 0};
    for (int tm = 0; tm < 2; ++tm)
        for (int tn = 0; tn < 4; ++tn)
            for (int v = 0; v < 8; ++v) {
                int row = wm * 32 + tm * 16 + v + ((lane >> 4) << 3);
                acc[tm][tn][v] += b_out[row];
                float val = acc[tm][tn][v];
                cs[tn] += val;
                cq[tn] += val * val;
            }
    for (int tn = 0; tn < 4; ++tn) {     // lane & lane^16 hold complementary rows
        cs[tn] += __shfl_xor(cs[tn], 16, 32);
        cq[tn] += __shfl_xor(cq[tn], 16, 32);
    }
    if (lane < 16)
        for (int tn = 0; tn < 4; ++tn) {
            int c = wn * 64 + tn * 16 + lane;
            pSum[wm][c] = cs[tn];
            pSq[wm][c]  = cq[tn];
        }
    __syncthreads();
    if (tid < 128) {
        float s = pSum[0][tid] + pSum[1][tid] + pSum[2][tid] + pSum[3][tid];
        float q = pSq[0][tid] + pSq[1][tid] + pSq[2][tid] + pSq[3][tid];
        float mean = s * (1.0f / 128.0f);
        float var  = q * (1.0f / 128.0f) - mean * mean;
        sMean[tid] = mean;
        sRstd[tid] = rsqrtf(var + 1e-5f);
    }
    __syncthreads();

    for (int tm = 0; tm < 2; ++tm)
        for (int tn = 0; tn < 4; ++tn)
            for (int v = 0; v < 8; ++v) {
                int row = wm * 32 + tm * 16 + v + ((lane >> 4) << 3);
                int cl  = wn * 64 + tn * 16 + (lane & 15);
                float val = (acc[tm][tn][v] - sMean[cl]) * sRstd[cl] * g[row];
                y[(size_t)(bt * 128 + row) * 1024 + n0 + cl] = val;
            }
}

extern "C" void kernel_launch(void* const* d_in, const int* in_sizes, int n_in,
                              void* d_out, int out_size, void* d_ws, size_t ws_size,
                              hipStream_t stream) {
    const float* x     = (const float*)d_in[0];   // (4,32,128,32,32)
    const float* w_qkv = (const float*)d_in[1];   // (768,128)
    const float* w_out = (const float*)d_in[2];   // (128,256)
    const float* b_out = (const float*)d_in[3];   // (128,)
    const float* g     = (const float*)d_in[4];   // (128,)
    float* y = (float*)d_out;

    unsigned short* qkv = (unsigned short*)d_ws;                   // 128*768*1024 bf16
    unsigned short* att = qkv + (size_t)128 * 768 * 1024;          // 128*256*1024 bf16

    qkv_gemm  <<<dim3(8, 6, 128), 256, 0, stream>>>(x, w_qkv, qkv);
    attn_core <<<dim3(256, 32, 1), 128, 0, stream>>>(qkv, att);
    out_gemm_ln<<<dim3(8, 128, 1), 256, 0, stream>>>(att, w_out, b_out, g, y);
}
